// MHA_pro_37864431682659
// MI455X (gfx1250) — compile-verified, matrix operand order checked by reading, not run on hardware
//
#include <hip/hip_runtime.h>
#include <hip/hip_bf16.h>
#include <math.h>

// ---------------- problem constants ----------------
enum : int { BB = 2, TT = 1024, CC = 1024, NH = 16, HS = 64, ROT = 32 };

typedef __attribute__((ext_vector_type(16))) __bf16 v16bf;
typedef __attribute__((ext_vector_type(8)))  __bf16 v8bf;
typedef __attribute__((ext_vector_type(8)))  float  v8f;

#define WMMA_BF16(a, b, c) \
  __builtin_amdgcn_wmma_f32_16x16x32_bf16(false, (a), false, (b), (short)0, (c), false, false)

// Fragment loader: lane supplies base = row_ptr + kb (kb = 8*(lane>=16)).
// Elements 0..7  <- K = kb+0..7   (16B load)
// Elements 8..15 <- K = kb+16..23 (16B load)
static __device__ __forceinline__ v16bf load_frag(const __bf16* p) {
  v8bf lo = *(const v8bf*)p;
  v8bf hi = *(const v8bf*)(p + 16);
  v16bf r;
#pragma unroll
  for (int i = 0; i < 8; ++i) { r[i] = lo[i]; r[i + 8] = hi[i]; }
  return r;
}

// ---------------- 1) time-shift + f32->bf16 ----------------
__global__ void k_shift(const float* __restrict__ x, __bf16* __restrict__ xs, int n) {
  int i = blockIdx.x * blockDim.x + threadIdx.x;
  if (i >= n) return;
  int c = i % CC, t = (i / CC) % TT, b = i / (CC * TT);
  float v;
  if (c < CC / 2) v = (t > 0) ? x[((size_t)b * TT + t - 1) * CC + c] : 0.f;
  else            v = x[((size_t)b * TT + t) * CC + c];
  xs[i] = (__bf16)v;
}

// ---------------- 2) W[k][n] f32 -> Wt[n][k] bf16 ----------------
__global__ void k_convT(const float* __restrict__ W, __bf16* __restrict__ Wt, int K, int N) {
  int i = blockIdx.x * blockDim.x + threadIdx.x;
  if (i >= K * N) return;
  int k = i % K, n = i / K;
  Wt[i] = (__bf16)W[(size_t)k * N + n];
}

// ---------------- generic bf16 WMMA GEMM, 2x2 register-blocked (32x32 per wave) ----------------
// out = A @ Bt^T + bias (opt *gamma[m%T]); 4 fragment loads feed 4 WMMAs per K-step.
__global__ void k_gemm(const __bf16* __restrict__ A, const __bf16* __restrict__ Bt,
                       const float* __restrict__ bias, const float* __restrict__ gamma,
                       float* __restrict__ out, int M, int N, int K, int T) {
  int wid  = (blockIdx.x * blockDim.x + threadIdx.x) >> 5;
  int lane = threadIdx.x & 31;
  int hf = lane >> 4, ml = lane & 15;
  int ntn = N >> 5;                          // 32-wide N tiles
  int tm = wid / ntn, tn = wid % ntn;
  if (tm >= (M >> 5)) return;
  int m0 = tm * 32, n0 = tn * 32;
  const __bf16* pa0 = A  + (size_t)(m0 + ml)      * K + hf * 8;
  const __bf16* pa1 = A  + (size_t)(m0 + 16 + ml) * K + hf * 8;
  const __bf16* pb0 = Bt + (size_t)(n0 + ml)      * K + hf * 8;
  const __bf16* pb1 = Bt + (size_t)(n0 + 16 + ml) * K + hf * 8;
  v8f a00 = {}, a01 = {}, a10 = {}, a11 = {};
  for (int k0 = 0; k0 < K; k0 += 32) {
    v16bf fa0 = load_frag(pa0 + k0);
    v16bf fa1 = load_frag(pa1 + k0);
    v16bf fb0 = load_frag(pb0 + k0);
    v16bf fb1 = load_frag(pb1 + k0);
    a00 = WMMA_BF16(fa0, fb0, a00);
    a01 = WMMA_BF16(fa0, fb1, a01);
    a10 = WMMA_BF16(fa1, fb0, a10);
    a11 = WMMA_BF16(fa1, fb1, a11);
  }
#pragma unroll
  for (int bm = 0; bm < 2; ++bm)
#pragma unroll
    for (int bn = 0; bn < 2; ++bn) {
      const v8f& acc = bm ? (bn ? a11 : a10) : (bn ? a01 : a00);
      int n = n0 + bn * 16 + ml;
      float bv = bias ? bias[n] : 0.f;
#pragma unroll
      for (int r = 0; r < 8; ++r) {
        int m = m0 + bm * 16 + r + hf * 8;
        float v = acc[r] + bv;
        if (gamma) v *= gamma[m % T];
        out[(size_t)m * N + n] = v;
      }
    }
}

// ---------------- 4) RoPE + relayout to WMMA-friendly buffers ----------------
// Qh/Kh: [b*16+h][t][64] bf16 ;  Vt: [b*16+h][d][1024] bf16 (transposed for att@v B-frags)
__global__ void k_rope(const float* __restrict__ Q, const float* __restrict__ Kr,
                       const float* __restrict__ V, __bf16* __restrict__ Qh,
                       __bf16* __restrict__ Kh, __bf16* __restrict__ Vt, int n) {
  int i = blockIdx.x * blockDim.x + threadIdx.x;
  if (i >= n) return;
  int nn = i % CC, t = (i / CC) % TT, b = i / (CC * TT);
  int h = nn >> 6, d = nn & 63;
  float qv = Q[i], kv = Kr[i];
  if (d < ROT) {
    int fi = d & 15;
    float fr = (float)t * __powf(1024.0f, -(float)fi / 16.0f);
    float cs = __cosf(fr), sn = __sinf(fr);
    int pd = (d < 16) ? d + 16 : d - 16;
    float sgn = (d < 16) ? -1.f : 1.f;
    size_t base = (size_t)i - d;
    qv = qv * cs + sgn * Q[base + pd] * sn;
    kv = kv * cs + sgn * Kr[base + pd] * sn;
  }
  int bh = b * NH + h;
  Qh[((size_t)bh * TT + t) * HS + d] = (__bf16)qv;
  Kh[((size_t)bh * TT + t) * HS + d] = (__bf16)kv;
  Vt[((size_t)bh * HS + d) * TT + t] = (__bf16)V[i];
}

// ---------------- 5) softmax stats over ALL s (max & sum) ----------------
__global__ void k_stats(const __bf16* __restrict__ Qh, const __bf16* __restrict__ Kh,
                        float* __restrict__ Mb, float* __restrict__ Zb) {
  int wid  = (blockIdx.x * blockDim.x + threadIdx.x) >> 5;
  int lane = threadIdx.x & 31;
  int hf = lane >> 4, ml = lane & 15;
  int tile = wid & 63, bh = wid >> 6;      // 2048 waves: [b*16+h][t-tile]
  int t0 = tile * 16;
  const __bf16* qrow = Qh + ((size_t)bh * TT + t0 + ml) * HS + hf * 8;
  v16bf qa0 = load_frag(qrow);
  v16bf qa1 = load_frag(qrow + 32);
  float mr[8], zz[8];
#pragma unroll
  for (int r = 0; r < 8; ++r) { mr[r] = -3.0e38f; zz[r] = 0.f; }
  for (int st = 0; st < TT / 16; ++st) {
    const __bf16* krow = Kh + ((size_t)bh * TT + st * 16 + ml) * HS + hf * 8;
    v8f s = {};
    s = WMMA_BF16(qa0, load_frag(krow), s);
    s = WMMA_BF16(qa1, load_frag(krow + 32), s);
#pragma unroll
    for (int r = 0; r < 8; ++r) {
      float x = s[r] * 0.125f;               // 1/sqrt(64)
      float tmax = x;
      tmax = fmaxf(tmax, __shfl_xor(tmax, 8, 16));
      tmax = fmaxf(tmax, __shfl_xor(tmax, 4, 16));
      tmax = fmaxf(tmax, __shfl_xor(tmax, 2, 16));
      tmax = fmaxf(tmax, __shfl_xor(tmax, 1, 16));
      float mnew = fmaxf(mr[r], tmax);
      float e = __expf(x - mnew);
      e += __shfl_xor(e, 8, 16);
      e += __shfl_xor(e, 4, 16);
      e += __shfl_xor(e, 2, 16);
      e += __shfl_xor(e, 1, 16);
      zz[r] = zz[r] * __expf(mr[r] - mnew) + e;
      mr[r] = mnew;
    }
  }
  if (ml == 0) {
#pragma unroll
    for (int r = 0; r < 8; ++r) {
      int t = t0 + r + hf * 8;
      Mb[(size_t)bh * TT + t] = mr[r];
      Zb[(size_t)bh * TT + t] = zz[r];
    }
  }
}

// ---------------- 6) fused attention: scores -> p*w -> Wmix -> @v  (att stays on-WGP) ----------------
__global__ void __launch_bounds__(512)
k_attn(const __bf16* __restrict__ Qh, const __bf16* __restrict__ Kh,
       const __bf16* __restrict__ Vt, const float* __restrict__ Mb,
       const float* __restrict__ Zb, const float* __restrict__ tw,
       const float* __restrict__ alpha, const float* __restrict__ beta,
       const float* __restrict__ Wmix, __bf16* __restrict__ Y) {
  __shared__ float pw[NH * 16 * 32];                   // 32 KB of 320 KB LDS
  int tile = blockIdx.x & 63, b = blockIdx.x >> 6;
  int wave = threadIdx.x >> 5;                         // = input head h and output head g
  int lane = threadIdx.x & 31;
  int hf = lane >> 4, ml = lane & 15;
  int t0 = tile * 16;
  int h = wave, g = wave;
  int bh = b * NH + h, bg = b * NH + g;

  const __bf16* qrow = Qh + ((size_t)bh * TT + t0 + ml) * HS + hf * 8;
  v16bf qa0 = load_frag(qrow);
  v16bf qa1 = load_frag(qrow + 32);

  float mr[8], zi[8], bt[8];
#pragma unroll
  for (int r = 0; r < 8; ++r) {
    int t = t0 + r + hf * 8;
    mr[r] = Mb[(size_t)bh * TT + t];
    zi[r] = 1.0f / Zb[(size_t)bh * TT + t];
    bt[r] = beta[h * TT + t];
  }
  float wrow[NH];
#pragma unroll
  for (int j = 0; j < NH; ++j) wrow[j] = Wmix[g * NH + j];

  v8f yac[4] = {};
  int nch = (t0 + 16 + 31) >> 5;
  for (int ch = 0; ch < nch; ++ch) {
    int s0 = ch * 32;
    // phase A: this wave = head h; two 16-wide s subtiles of exp/Z * w -> LDS
#pragma unroll
    for (int sub = 0; sub < 2; ++sub) {
      int sb = s0 + sub * 16;
      const __bf16* krow = Kh + ((size_t)bh * TT + sb + ml) * HS + hf * 8;
      v8f s = {};
      s = WMMA_BF16(qa0, load_frag(krow), s);
      s = WMMA_BF16(qa1, load_frag(krow + 32), s);
#pragma unroll
      for (int r = 0; r < 8; ++r) {
        int tl = r + hf * 8, t = t0 + tl, sg = sb + ml;
        float p = __expf(s[r] * 0.125f - mr[r]) * zi[r];
        float wv = 0.f;
        if (sg <= t)
          wv = tw[h * TT + (TT - 1) - (t - sg)] * alpha[h * TT + sg] * bt[r];
        pw[(h * 16 + tl) * 32 + sub * 16 + ml] = p * wv;
      }
    }
    __syncthreads();
    // phase B: this wave = output head g; mix 16 heads, repack to A-frag, 4 WMMAs vs Vt
    v16bf am;
#pragma unroll
    for (int j = 0; j < 8; ++j) {
      int sl = hf * 8 + ((j < 4) ? 2 * j : 16 + 2 * (j - 4));
      float a0 = 0.f, a1 = 0.f;
#pragma unroll
      for (int hh = 0; hh < NH; ++hh) {
        const float2 v2 = *(const float2*)&pw[(hh * 16 + ml) * 32 + sl];
        a0 += wrow[hh] * v2.x;
        a1 += wrow[hh] * v2.y;
      }
      am[2 * j]     = (__bf16)a0;
      am[2 * j + 1] = (__bf16)a1;
    }
#pragma unroll
    for (int dt = 0; dt < 4; ++dt) {
      const __bf16* vrow = Vt + ((size_t)bg * HS + dt * 16 + ml) * TT + s0 + hf * 8;
      yac[dt] = WMMA_BF16(am, load_frag(vrow), yac[dt]);
    }
    __syncthreads();
  }
  // write y[b][t][g*64+d] as bf16 for the final WMMA GEMM
#pragma unroll
  for (int dt = 0; dt < 4; ++dt)
#pragma unroll
    for (int r = 0; r < 8; ++r) {
      int t = t0 + r + hf * 8;
      int n = g * HS + dt * 16 + ml;
      Y[((size_t)b * TT + t) * CC + n] = (__bf16)yac[dt][r];
    }
}

// ---------------- host-side orchestration ----------------
extern "C" void kernel_launch(void* const* d_in, const int* in_sizes, int n_in,
                              void* d_out, int out_size, void* d_ws, size_t ws_size,
                              hipStream_t stream) {
  const float* x     = (const float*)d_in[0];
  const float* tw    = (const float*)d_in[1];
  const float* alpha = (const float*)d_in[2];
  const float* beta  = (const float*)d_in[3];
  const float* gamma = (const float*)d_in[4];
  const float* Wq = (const float*)d_in[5];  const float* bq = (const float*)d_in[6];
  const float* Wk = (const float*)d_in[7];  const float* bk = (const float*)d_in[8];
  const float* Wv = (const float*)d_in[9];  const float* bv = (const float*)d_in[10];
  const float* Wmix = (const float*)d_in[11];
  const float* Wo = (const float*)d_in[12]; const float* bo = (const float*)d_in[13];
  float* out = (float*)d_out;

  char* ws = (char*)d_ws;
  const size_t MB = 1u << 20;
  __bf16* XS   = (__bf16*)(ws + 0);          // 4 MB
  __bf16* WQT  = (__bf16*)(ws + 4 * MB);     // 2 MB each
  __bf16* WKT  = (__bf16*)(ws + 6 * MB);
  __bf16* WVT  = (__bf16*)(ws + 8 * MB);
  __bf16* WOT  = (__bf16*)(ws + 10 * MB);
  float*  QR   = (float*)(ws + 12 * MB);     // 8 MB each
  float*  KR   = (float*)(ws + 20 * MB);
  float*  VR   = (float*)(ws + 28 * MB);
  __bf16* QHB  = (__bf16*)(ws + 36 * MB);    // 4 MB each
  __bf16* KHB  = (__bf16*)(ws + 40 * MB);
  __bf16* VTB  = (__bf16*)(ws + 44 * MB);
  float*  MBUF = (float*)(ws + 48 * MB);     // 128 KB each
  float*  ZBUF = (float*)(ws + 48 * MB + (128u << 10));
  __bf16* YBF  = (__bf16*)(ws + 49 * MB);    // 4 MB

  const int nBTC = BB * TT * CC;             // 2M
  // 1) shift + bf16
  k_shift<<<(nBTC + 255) / 256, 256, 0, stream>>>(x, XS, nBTC);
  // 2) weight transpose-converts
  k_convT<<<(CC * CC + 255) / 256, 256, 0, stream>>>(Wq, WQT, CC, CC);
  k_convT<<<(CC * CC + 255) / 256, 256, 0, stream>>>(Wk, WKT, CC, CC);
  k_convT<<<(CC * CC + 255) / 256, 256, 0, stream>>>(Wv, WVT, CC, CC);
  k_convT<<<(CC * CC + 255) / 256, 256, 0, stream>>>(Wo, WOT, CC, CC);
  // 3) QKV GEMMs: (B*T=2048) x 1024 x 1024, 32x32 tile per wave, 8 waves per block
  const int M = BB * TT, N = CC, K = CC;
  int gblk = (M / 32) * (N / 32) / 8;        // 256 blocks
  k_gemm<<<gblk, 256, 0, stream>>>(XS, WQT, bq, nullptr, QR, M, N, K, TT);
  k_gemm<<<gblk, 256, 0, stream>>>(XS, WKT, bk, nullptr, KR, M, N, K, TT);
  k_gemm<<<gblk, 256, 0, stream>>>(XS, WVT, bv, nullptr, VR, M, N, K, TT);
  // 4) RoPE + relayout
  k_rope<<<(nBTC + 255) / 256, 256, 0, stream>>>(QR, KR, VR, QHB, KHB, VTB, nBTC);
  // 5) softmax stats: 2048 waves = 256 blocks x 8 waves
  k_stats<<<256, 256, 0, stream>>>(QHB, KHB, MBUF, ZBUF);
  // 6) fused attention: one block per (b, 16-row tile), 16 waves = 16 heads
  k_attn<<<BB * (TT / 16), 512, 0, stream>>>(QHB, KHB, VTB, MBUF, ZBUF,
                                             tw, alpha, beta, Wmix, YBF);
  // 7) output GEMM with bias + gamma epilogue
  k_gemm<<<gblk, 256, 0, stream>>>(YBF, WOT, bo, gamma, out, M, N, K, TT);
}
